// Head_8426725834935
// MI455X (gfx1250) — matrix operand order repaired and checked by reading, not compiled, not run on hardware
//
#include <hip/hip_runtime.h>

// Problem constants (match reference)
constexpr int B = 16;
constexpr int T = 1024;
constexpr int D = 768;
constexpr int H = 64;

typedef __attribute__((ext_vector_type(16))) __bf16 v16bf;
typedef __attribute__((ext_vector_type(8)))  float  v8f;
typedef __attribute__((ext_vector_type(4)))  unsigned int u32x4;
typedef __attribute__((ext_vector_type(8)))  int i32x8;
typedef __attribute__((ext_vector_type(4)))  int i32x4;

#if __has_builtin(__builtin_amdgcn_tensor_load_to_lds) && \
    __has_builtin(__builtin_amdgcn_s_wait_tensorcnt)
#define USE_TDM 1
#else
#define USE_TDM 0
#endif

union FragB {            // 8 VGPRs = 16 bf16 elements
    unsigned int u[8];
    v16bf v;
};

__device__ __forceinline__ v8f wmma_bf16(const v16bf& a, const v16bf& b, const v8f& c) {
    return __builtin_amdgcn_wmma_f32_16x16x32_bf16(
        /*neg_a=*/false, a, /*neg_b=*/false, b,
        /*c_mod=*/(short)0, c, /*reuse_a=*/false, /*reuse_b=*/false);
}

// ---------------------------------------------------------------------------
// Kernel 1: QKV projection.  q = (x @ Wq^T) * D^-0.5 ; k = x @ Wk^T ; v = x @ Wv^T
// Block = 128 threads (4 waves). Block handles 64 T-rows of one batch.
// Wave w handles rows [16w, 16w+16) of the tile; all 4 N-tiles (H=64) of q,k,v.
// ---------------------------------------------------------------------------
__global__ __launch_bounds__(128) void qkv_proj_kernel(
    const float* __restrict__ x,
    const float* __restrict__ Wq,
    const float* __restrict__ Wk,
    const float* __restrict__ Wv,
    __bf16* __restrict__ qo, __bf16* __restrict__ ko, __bf16* __restrict__ vo)
{
    __shared__ __bf16 xs[64][36];       // x tile [row][k], padded stride
    __shared__ __bf16 wsh[3][64][36];   // Wq/Wk/Wv tiles [h][k]

    const int tid  = threadIdx.x;
    const int lane = tid & 31;
    const int w    = tid >> 5;          // wave id 0..3
    const int g    = lane >> 4;         // half-wave group
    const int mn   = lane & 15;         // row (A) / col (B,C) index

    const int bb = blockIdx.x / (T / 64);
    const int t0 = (blockIdx.x % (T / 64)) * 64;

    const float* xrow = x + ((size_t)bb * T + t0) * D;
    const float* Wm[3] = {Wq, Wk, Wv};

    v8f accq[4], acck[4], accv[4];
#pragma unroll
    for (int nt = 0; nt < 4; ++nt) { accq[nt] = {}; acck[nt] = {}; accv[nt] = {}; }

    const int c4 = (tid & 7) * 4;       // 0..28
    const int r0 = tid >> 3;            // 0..15

    for (int k0 = 0; k0 < D; k0 += 32) {
        __syncthreads();
        // ---- stage x chunk [64 x 32] as bf16 ----
#pragma unroll
        for (int p = 0; p < 4; ++p) {
            int r = r0 + p * 16;
            float4 fv = *(const float4*)(xrow + (size_t)r * D + k0 + c4);
            struct bf4 { __bf16 a, b, c, d; };
            *(bf4*)&xs[r][c4] = bf4{(__bf16)fv.x, (__bf16)fv.y, (__bf16)fv.z, (__bf16)fv.w};
        }
        // ---- stage weight chunks [64 x 32] x 3 ----
#pragma unroll
        for (int m = 0; m < 3; ++m) {
#pragma unroll
            for (int p = 0; p < 4; ++p) {
                int r = r0 + p * 16;
                float4 fv = *(const float4*)(Wm[m] + (size_t)r * D + k0 + c4);
                struct bf4 { __bf16 a, b, c, d; };
                *(bf4*)&wsh[m][r][c4] = bf4{(__bf16)fv.x, (__bf16)fv.y, (__bf16)fv.z, (__bf16)fv.w};
            }
        }
        if (k0 + 32 < D)  // hint next chunk toward the caches
            __builtin_prefetch(xrow + (size_t)r0 * D + k0 + 32, 0, 1);
        __syncthreads();

        // ---- A fragment: 16x32 bf16, ISA layout (row = mn, paired K) ----
        FragB a;
        const int arow = 16 * w + mn;
#pragma unroll
        for (int j = 0; j < 8; ++j) {
            int kk = (j < 4) ? (8 * g + 2 * j) : (16 + 8 * g + 2 * (j - 4));
            a.u[j] = *(const unsigned int*)&xs[arow][kk];
        }
        // ---- B fragments + 12 WMMAs ----
#pragma unroll
        for (int nt = 0; nt < 4; ++nt) {
            const int hrow = nt * 16 + mn;
            FragB bq, bk, bv;
#pragma unroll
            for (int j = 0; j < 8; ++j) {
                int kk = (j < 4) ? (8 * g + 2 * j) : (16 + 8 * g + 2 * (j - 4));
                bq.u[j] = *(const unsigned int*)&wsh[0][hrow][kk];
                bk.u[j] = *(const unsigned int*)&wsh[1][hrow][kk];
                bv.u[j] = *(const unsigned int*)&wsh[2][hrow][kk];
            }
            accq[nt] = wmma_bf16(a.v, bq.v, accq[nt]);
            acck[nt] = wmma_bf16(a.v, bk.v, acck[nt]);
            accv[nt] = wmma_bf16(a.v, bv.v, accv[nt]);
        }
    }

    // ---- store q (pre-scaled by D^-0.5), k, v as bf16 ----
    const float qscale = 0.03608439182435161f; // 768^-0.5
#pragma unroll
    for (int nt = 0; nt < 4; ++nt) {
#pragma unroll
        for (int r = 0; r < 8; ++r) {
            int row = 16 * w + r + 8 * g;
            int col = nt * 16 + mn;
            size_t idx = ((size_t)bb * T + t0 + row) * H + col;
            qo[idx] = (__bf16)(accq[nt][r] * qscale);
            ko[idx] = (__bf16)acck[nt][r];
            vo[idx] = (__bf16)accv[nt][r];
        }
    }
}

// ---------------------------------------------------------------------------
// Kernel 2: causal flash attention.  out = softmax(q k^T) v   (scale already in q)
// Block = 128 threads (4 waves) -> 64 q rows of one batch. Wave w: rows qt..qt+15.
// K tile staged by the Tensor Data Mover (wave 0 issues, TENSORcnt-waited);
// V tile transposed through the VALU path. Key loop is block-uniform.
// ---------------------------------------------------------------------------
__global__ __launch_bounds__(128) void attn_kernel(
    const __bf16* __restrict__ q,
    const __bf16* __restrict__ k,
    const __bf16* __restrict__ v,
    float* __restrict__ out)
{
    __shared__ __bf16 kt[32][72];       // K chunk [s][h], 128B data + 16B pad per row
    __shared__ __bf16 vt[64][36];       // V chunk transposed [h][s]
    __shared__ __bf16 ps[4][16][36];    // per-wave P staging [row][s]

    const int tid  = threadIdx.x;
    const int lane = tid & 31;
    const int w    = tid >> 5;
    const int g    = lane >> 4;
    const int mn   = lane & 15;

    const int bb = blockIdx.x / (T / 64);
    const int t0 = (blockIdx.x % (T / 64)) * 64;
    const int qt = t0 + 16 * w;

    // ---- load Q A-fragments once (K=64 -> two K=32 fragments) ----
    FragB qa0, qa1;
    {
        const __bf16* qrow = q + ((size_t)bb * T + qt + mn) * H;
#pragma unroll
        for (int j = 0; j < 8; ++j) {
            int kk = (j < 4) ? (8 * g + 2 * j) : (16 + 8 * g + 2 * (j - 4));
            qa0.u[j] = *(const unsigned int*)&qrow[kk];
            qa1.u[j] = *(const unsigned int*)&qrow[kk + 32];
        }
    }

    v8f   O[4];
    float m_r[8], l_r[8];
#pragma unroll
    for (int nt = 0; nt < 4; ++nt) O[nt] = {};
#pragma unroll
    for (int r = 0; r < 8; ++r) { m_r[r] = -1e30f; l_r[r] = 0.0f; }

    const __bf16* kbase = k + (size_t)bb * T * H;
    const __bf16* vbase = v + (size_t)bb * T * H;

    const int nchunks = t0 / 32 + 2;    // covers diagonal row t0+63
    const int h4 = (tid & 15) * 4;      // loader: col group
    const int s0 = tid >> 4;            // loader: row group

    for (int ci = 0; ci < nchunks; ++ci) {
        const int c = ci * 32;
        __syncthreads();

#if USE_TDM
        // ---- TDM: stage K [32 x 64] bf16 tile into padded LDS (wave 0 only) ----
        if (w == 0) {
            const unsigned long long ga = (unsigned long long)(size_t)(kbase + (size_t)c * H);
            const unsigned lds_base = (unsigned)(size_t)&kt[0][0];
            u32x4 g0;
            g0[0] = 1u;                                   // count=1, user descriptor
            g0[1] = lds_base;                             // lds_addr
            g0[2] = (unsigned)ga;                         // global_addr[31:0]
            g0[3] = ((unsigned)(ga >> 32) & 0x01FFFFFFu)  // global_addr[56:32]
                    | 0x80000000u;                        // type=2 ("image")
            i32x8 g1;
            g1[0] = (int)((1u << 16)      // data_size = 2 bytes
                        | (1u << 20)      // pad_enable
                        | (4u << 22)      // pad_interval: 32 DWORDs (=128B row)
                        | (3u << 25));    // pad_amount:   4 DWORDs (=16B pad)
            g1[1] = (int)(64u << 16);     // tensor_dim0 = 64 (low 16 bits)
            g1[2] = (int)(32u << 16);     // tensor_dim1 = 32 (low 16 bits)
            g1[3] = (int)(64u << 16);     // tile_dim0 = 64
            g1[4] = 32;                   // tile_dim1 = 32 (tile_dim2 = 0)
            g1[5] = 64;                   // tensor_dim0_stride = H = 64
            g1[6] = 0;
            g1[7] = 0;
            i32x4 z4 = {};
            i32x8 z8 = {};
            __builtin_amdgcn_tensor_load_to_lds(g0, g1, z4, z4, z8, 0);
        }
#endif
        // ---- stage V^T [64 x 32] (and K fallback) ----
#pragma unroll
        for (int p = 0; p < 4; ++p) {
            int s = s0 + p * 8;
#if !USE_TDM
            uint2 kd = *(const uint2*)(kbase + (size_t)(c + s) * H + h4);
            *(uint2*)&kt[s][h4] = kd;
#endif
            union { uint2 u; __bf16 e[4]; } cv;
            cv.u = *(const uint2*)(vbase + (size_t)(c + s) * H + h4);
            vt[h4 + 0][s] = cv.e[0];
            vt[h4 + 1][s] = cv.e[1];
            vt[h4 + 2][s] = cv.e[2];
            vt[h4 + 3][s] = cv.e[3];
        }
#if USE_TDM
        if (w == 0) __builtin_amdgcn_s_wait_tensorcnt(0);  // K tile in LDS
#endif
        __syncthreads();

        if (c > qt + 15) continue;      // fully masked for this wave (uniform per wave)

        // ---- scores: S[16 x 32] = Q(16x64) @ K^T, two N-tiles ----
        v8f S[2];
#pragma unroll
        for (int st = 0; st < 2; ++st) {
            FragB kb0, kb1;
            const int srow = st * 16 + mn;
#pragma unroll
            for (int j = 0; j < 8; ++j) {
                int kk = (j < 4) ? (8 * g + 2 * j) : (16 + 8 * g + 2 * (j - 4));
                kb0.u[j] = *(const unsigned int*)&kt[srow][kk];
                kb1.u[j] = *(const unsigned int*)&kt[srow][kk + 32];
            }
            v8f acc = {};
            acc = wmma_bf16(qa0.v, kb0.v, acc);
            acc = wmma_bf16(qa1.v, kb1.v, acc);
            S[st] = acc;
        }

        // ---- causal mask (only chunks crossing the diagonal) ----
        if (c + 31 > qt) {
#pragma unroll
            for (int st = 0; st < 2; ++st) {
#pragma unroll
                for (int r = 0; r < 8; ++r) {
                    int col = c + st * 16 + mn;
                    int row = qt + r + 8 * g;
                    if (col > row) S[st][r] = -1e30f;
                }
            }
        }

        // ---- online softmax update ----
        float alpha[8];
#pragma unroll
        for (int r = 0; r < 8; ++r) {
            float mv = fmaxf(S[0][r], S[1][r]);
            mv = fmaxf(mv, __shfl_xor(mv, 1, 32));
            mv = fmaxf(mv, __shfl_xor(mv, 2, 32));
            mv = fmaxf(mv, __shfl_xor(mv, 4, 32));
            mv = fmaxf(mv, __shfl_xor(mv, 8, 32));
            float mnew = fmaxf(m_r[r], mv);
            float a    = __expf(m_r[r] - mnew);
            float p0   = __expf(S[0][r] - mnew);
            float p1   = __expf(S[1][r] - mnew);
            S[0][r] = p0; S[1][r] = p1;
            float sum = p0 + p1;
            sum += __shfl_xor(sum, 1, 32);
            sum += __shfl_xor(sum, 2, 32);
            sum += __shfl_xor(sum, 4, 32);
            sum += __shfl_xor(sum, 8, 32);
            l_r[r] = l_r[r] * a + sum;
            m_r[r] = mnew;
            alpha[r] = a;
        }
#pragma unroll
        for (int nt = 0; nt < 4; ++nt)
#pragma unroll
            for (int r = 0; r < 8; ++r) O[nt][r] *= alpha[r];

        // ---- C-layout -> A-layout for P via per-wave LDS staging ----
#pragma unroll
        for (int st = 0; st < 2; ++st)
#pragma unroll
            for (int r = 0; r < 8; ++r)
                ps[w][r + 8 * g][st * 16 + mn] = (__bf16)S[st][r];

        FragB pa;
#pragma unroll
        for (int j = 0; j < 8; ++j) {
            int kk = (j < 4) ? (8 * g + 2 * j) : (16 + 8 * g + 2 * (j - 4));
            pa.u[j] = *(const unsigned int*)&ps[w][mn][kk];
        }

        // ---- O += P @ V  (4 N-tiles over H=64) ----
#pragma unroll
        for (int nt = 0; nt < 4; ++nt) {
            FragB vb;
            const int hrow = nt * 16 + mn;
#pragma unroll
            for (int j = 0; j < 8; ++j) {
                int kk = (j < 4) ? (8 * g + 2 * j) : (16 + 8 * g + 2 * (j - 4));
                vb.u[j] = *(const unsigned int*)&vt[hrow][kk];
            }
            O[nt] = wmma_bf16(pa.v, vb.v, O[nt]);
        }
    }

    // ---- epilogue: normalize and store fp32 ----
#pragma unroll
    for (int r = 0; r < 8; ++r) {
        float inv = 1.0f / l_r[r];
        int row = qt + r + 8 * g;
#pragma unroll
        for (int nt = 0; nt < 4; ++nt) {
            out[((size_t)bb * T + row) * H + nt * 16 + mn] = O[nt][r] * inv;
        }
    }
}

// ---------------------------------------------------------------------------
extern "C" void kernel_launch(void* const* d_in, const int* in_sizes, int n_in,
                              void* d_out, int out_size, void* d_ws, size_t ws_size,
                              hipStream_t stream)
{
    (void)in_sizes; (void)n_in; (void)out_size; (void)ws_size;
    const float* x  = (const float*)d_in[0];
    const float* Wq = (const float*)d_in[1];
    const float* Wk = (const float*)d_in[2];
    const float* Wv = (const float*)d_in[3];
    float* out = (float*)d_out;

    __bf16* qb = (__bf16*)d_ws;                    // B*T*H bf16 each
    __bf16* kb = qb + (size_t)B * T * H;
    __bf16* vb = kb + (size_t)B * T * H;

    dim3 grid(B * (T / 64));
    dim3 block(128);
    qkv_proj_kernel<<<grid, block, 0, stream>>>(x, Wq, Wk, Wv, qb, kb, vb);
    attn_kernel<<<grid, block, 0, stream>>>(qb, kb, vb, out);
}
